// Qwen3MoeSparseMLPBlock_53042846105769
// MI455X (gfx1250) — matrix operand order repaired and checked by reading, not compiled, not run on hardware
//
#include <hip/hip_runtime.h>
#include <hip/hip_bf16.h>

#define T_TOK 2048
#define H_DIM 2048
#define I_DIM 768
#define TWO_I 1536
#define N_EXP 16
#define TOPK  4
#define TILE_M 64

typedef __attribute__((ext_vector_type(16))) __bf16 v16bf;
typedef __attribute__((ext_vector_type(8)))  __bf16 v8bf;
typedef __attribute__((ext_vector_type(4)))  __bf16 v4bf;
typedef __attribute__((ext_vector_type(8)))  float  v8f;
typedef __attribute__((ext_vector_type(4)))  float  v4f;

// ---- workspace layout (bytes), total ~227.5 MB ----
#define WS_XBF   0ull                 // x bf16:                  8,388,608
#define WS_WGUT  8388608ull           // Wgu^T bf16 [E][2I][H]: 100,663,296
#define WS_WDT   109051904ull         // Wd^T  bf16 [E][H][I]:   50,331,648
#define WS_YB    159383552ull         // y_buf f32 [T*4][H]:     67,108,864
#define WS_CNT   226492416ull         // counts (16 ints, padded 256B)
#define WS_TOK   226492672ull         // packed (t*4+j) ids E*T:    524,288
#define WS_TW    227016960ull         // routing weights E*T:       524,288
#define WS_NEED  227541248ull

__global__ __launch_bounds__(32) void k_zero_cnt(int* __restrict__ counts) {
  if (threadIdx.x < N_EXP) counts[threadIdx.x] = 0;
}

__global__ __launch_bounds__(256) void k_cvt(const float* __restrict__ x,
                                             __bf16* __restrict__ xb) {
  size_t i = ((size_t)blockIdx.x * blockDim.x + threadIdx.x) * 4;
  v4f v = *(const v4f*)(x + i);
  v4bf b;
  b[0] = (__bf16)v[0]; b[1] = (__bf16)v[1];
  b[2] = (__bf16)v[2]; b[3] = (__bf16)v[3];
  *(v4bf*)(xb + i) = b;
}

// Transpose-convert per expert: src f32 [R][C] -> dst bf16 [C][R]. grid=(C/32,R/32,E)
__global__ __launch_bounds__(256) void k_tconv(const float* __restrict__ src,
                                               __bf16* __restrict__ dst,
                                               int R, int C) {
  __shared__ float t[32][33];
  size_t eoff = (size_t)blockIdx.z * R * C;
  src += eoff; dst += eoff;
  int c0 = blockIdx.x * 32, r0 = blockIdx.y * 32;
  int tc = threadIdx.x & 31, tr = threadIdx.x >> 5;
  #pragma unroll
  for (int i = 0; i < 4; ++i)
    t[tr + i * 8][tc] = src[(size_t)(r0 + tr + i * 8) * C + c0 + tc];
  __syncthreads();
  #pragma unroll
  for (int i = 0; i < 4; ++i)
    dst[(size_t)(c0 + tr + i * 8) * R + r0 + tc] = (__bf16)t[tc][tr + i * 8];
}

// Router: block = 256 threads = 16 tokens x 16 experts. List entries pack (t*4 + topk_slot).
__global__ __launch_bounds__(256) void k_router(const float* __restrict__ x,
                                                const float* __restrict__ gw,
                                                const float* __restrict__ gb,
                                                int* __restrict__ counts,
                                                int* __restrict__ tok_id,
                                                float* __restrict__ tok_w) {
  __shared__ float s_log[16][17];
  int tid = threadIdx.x;
  int tl = tid >> 4, e = tid & 15;
  int t = blockIdx.x * 16 + tl;
  float acc = gb[e];
  const float* xr = x + (size_t)t * H_DIM;
  #pragma unroll 8
  for (int h = 0; h < H_DIM; ++h)
    acc += xr[h] * gw[h * N_EXP + e];
  s_log[tl][e] = acc;
  __syncthreads();
  if (tid < 16) {
    float l[16];
    #pragma unroll
    for (int j = 0; j < 16; ++j) l[j] = s_log[tid][j];
    float m = l[0];
    #pragma unroll
    for (int j = 1; j < 16; ++j) m = fmaxf(m, l[j]);
    int id[TOPK]; float w[TOPK]; float s = 0.f;
    #pragma unroll
    for (int j = 0; j < TOPK; ++j) {
      float best = -3.4e38f; int bi = 0;
      #pragma unroll
      for (int q = 0; q < 16; ++q)
        if (l[q] > best) { best = l[q]; bi = q; }
      l[bi] = -3.4e38f;
      id[j] = bi;
      w[j] = __expf(best - m);   // softmax denom cancels after top-k renorm
      s += w[j];
    }
    float inv = 1.f / s;
    int tt = blockIdx.x * 16 + tid;
    #pragma unroll
    for (int j = 0; j < TOPK; ++j) {
      int slot = atomicAdd(&counts[id[j]], 1);
      tok_id[id[j] * T_TOK + slot] = tt * TOPK + j;   // unique y_buf row
      tok_w [id[j] * T_TOK + slot] = w[j] * inv;
    }
  }
}

__device__ __forceinline__ v8f wmma_bf16(v16bf a, v16bf b, v8f c) {
  return __builtin_amdgcn_wmma_f32_16x16x32_bf16(
      false, a, false, b, (short)0, c, false, false);
}

// A-fragment, contiguous K: lane<16 holds K={0..15}, lane>=16 K={16..31} (matches B).
template <typename P>
__device__ __forceinline__ v16bf load_a(P p) {
  return *(const v16bf*)(p);
}

__device__ __forceinline__ float silu_mul(float g, float u) {
  // fast reciprocal: v_rcp_f32 instead of IEEE divide expansion
  return g * __builtin_amdgcn_rcpf(1.f + __expf(-g)) * u;
}

// Fused expert MLP: grid = E * (T/TILE_M), 256 threads = 8 waves.
__global__ __launch_bounds__(256) void k_moe_mlp(const __bf16* __restrict__ xb,
                                                 const __bf16* __restrict__ Wgut, // [E][2I][H]
                                                 const __bf16* __restrict__ Wdt,  // [E][H][I]
                                                 const int* __restrict__ counts,
                                                 const int* __restrict__ tok_id,
                                                 const float* __restrict__ tok_w,
                                                 float* __restrict__ y_buf) {     // [T*4][H]
  __shared__ __align__(32) __bf16 s_h[TILE_M][784];   // 100,352 B
  __shared__ int   s_tok[TILE_M];
  __shared__ float s_w[TILE_M];
  __shared__ int   s_vld[TILE_M];

  int e    = blockIdx.x >> 5;               // 32 tiles per expert
  int tile = blockIdx.x & 31;
  int m0   = tile * TILE_M;
  int cnt  = counts[e];
  if (m0 >= cnt) return;

  int tid = threadIdx.x;
  if (tid < TILE_M) {
    int idx = m0 + tid;
    int v   = idx < cnt;
    int c   = v ? idx : (cnt - 1);
#if __has_builtin(__builtin_amdgcn_global_load_async_to_lds_b32)
    // per-lane async gather of the token list into LDS (ASYNCcnt-tracked)
    __builtin_amdgcn_global_load_async_to_lds_b32(
        (__attribute__((address_space(1))) int*)(tok_id + e * T_TOK + c),
        (__attribute__((address_space(3))) int*)&s_tok[tid], 0, 0);
#else
    s_tok[tid] = tok_id[e * T_TOK + c];
#endif
    s_w[tid]   = v ? tok_w[e * T_TOK + idx] : 0.f;
    s_vld[tid] = v;                         // padded duplicates must not store
  }
#if __has_builtin(__builtin_amdgcn_s_wait_asynccnt)
  __builtin_amdgcn_s_wait_asynccnt(0);
#else
  asm volatile("s_wait_asynccnt 0" ::: "memory");
#endif
  __syncthreads();

  const int lane = tid & 31, wave = tid >> 5;
  const int lane16 = lane & 15, khalf = lane >> 4;

  const __bf16* ap[4];
  #pragma unroll
  for (int m = 0; m < 4; ++m)
    ap[m] = xb + (size_t)(s_tok[m * 16 + lane16] >> 2) * H_DIM + khalf * 16;

  // ---------- Phase A: GU = X @ Wgu[e]; h = silu(g)*u -> s_h ----------
  const __bf16* wgut_e = Wgut + (size_t)e * TWO_I * H_DIM;
  for (int gp = wave; gp < I_DIM / 32; gp += 8) {     // 24 tile-pairs / 8 waves
    int gt0 = gp * 2;                                 // two adjacent 16-col tiles
    v8f ag0[4] = {{},{},{},{}}, ag1[4] = {{},{},{},{}};
    v8f au0[4] = {{},{},{},{}}, au1[4] = {{},{},{},{}};
    const __bf16* bg_p = wgut_e + (size_t)(gt0 * 16 + lane16) * H_DIM + khalf * 16;
    const __bf16* bu_p = wgut_e + (size_t)(I_DIM + gt0 * 16 + lane16) * H_DIM + khalf * 16;
    const size_t nstep = (size_t)16 * H_DIM;          // +16 columns
    for (int k0 = 0; k0 < H_DIM; k0 += 32) {
      __builtin_prefetch(bg_p + k0 + 256, 0, 1);
      __builtin_prefetch(bu_p + k0 + 256, 0, 1);
      v16bf bg0 = *(const v16bf*)(bg_p + k0);
      v16bf bg1 = *(const v16bf*)(bg_p + nstep + k0);
      v16bf bu0 = *(const v16bf*)(bu_p + k0);
      v16bf bu1 = *(const v16bf*)(bu_p + nstep + k0);
      #pragma unroll
      for (int m = 0; m < 4; ++m) {
        v16bf a = load_a(ap[m] + k0);
        ag0[m] = wmma_bf16(a, bg0, ag0[m]);
        ag1[m] = wmma_bf16(a, bg1, ag1[m]);
        au0[m] = wmma_bf16(a, bu0, au0[m]);
        au1[m] = wmma_bf16(a, bu1, au1[m]);
      }
    }
    #pragma unroll
    for (int m = 0; m < 4; ++m) {
      #pragma unroll
      for (int r = 0; r < 8; ++r) {                   // C layout: M = r + 8*khalf
        int row = m * 16 + r + khalf * 8;
        s_h[row][gt0 * 16 + lane16]      = (__bf16)silu_mul(ag0[m][r], au0[m][r]);
        s_h[row][gt0 * 16 + 16 + lane16] = (__bf16)silu_mul(ag1[m][r], au1[m][r]);
      }
    }
  }
  __syncthreads();

  // ---------- Phase B: Y = h @ Wd[e]; weighted scatter to y_buf ----------
  const __bf16* wdt_e = Wdt + (size_t)e * H_DIM * I_DIM;
  for (int np = wave; np < H_DIM / 32; np += 8) {     // 64 tile-pairs / 8 waves
    int n0 = np * 32;
    v8f ac0[4] = {{},{},{},{}}, ac1[4] = {{},{},{},{}};
    const __bf16* b_p = wdt_e + (size_t)(n0 + lane16) * I_DIM + khalf * 16;
    const size_t nstep = (size_t)16 * I_DIM;
    for (int k0 = 0; k0 < I_DIM; k0 += 32) {
      __builtin_prefetch(b_p + k0 + 256, 0, 1);
      v16bf b0 = *(const v16bf*)(b_p + k0);
      v16bf b1 = *(const v16bf*)(b_p + nstep + k0);
      #pragma unroll
      for (int m = 0; m < 4; ++m) {
        v16bf a = load_a(&s_h[m * 16 + lane16][k0 + khalf * 16]);
        ac0[m] = wmma_bf16(a, b0, ac0[m]);
        ac1[m] = wmma_bf16(a, b1, ac1[m]);
      }
    }
    #pragma unroll
    for (int m = 0; m < 4; ++m) {
      #pragma unroll
      for (int r = 0; r < 8; ++r) {
        int row = m * 16 + r + khalf * 8;
        if (s_vld[row]) {
          float* dst = y_buf + (size_t)s_tok[row] * H_DIM + n0 + lane16;
          dst[0]  = ac0[m][r] * s_w[row];
          dst[16] = ac1[m][r] * s_w[row];
        }
      }
    }
  }
}

// out[t][h] = sum_{j<4} y_buf[t*4+j][h]
__global__ __launch_bounds__(256) void k_comb(const float* __restrict__ yb,
                                              float* __restrict__ out) {
  size_t i = ((size_t)blockIdx.x * blockDim.x + threadIdx.x) * 4;
  size_t t = i / H_DIM;
  size_t h = i - t * H_DIM;
  const float* base = yb + (t * TOPK) * (size_t)H_DIM + h;
  v4f s0 = *(const v4f*)(base);
  v4f s1 = *(const v4f*)(base + H_DIM);
  v4f s2 = *(const v4f*)(base + 2 * H_DIM);
  v4f s3 = *(const v4f*)(base + 3 * H_DIM);
  v4f r = (s0 + s1) + (s2 + s3);
  *(v4f*)(out + i) = r;
}

extern "C" void kernel_launch(void* const* d_in, const int* in_sizes, int n_in,
                              void* d_out, int out_size, void* d_ws, size_t ws_size,
                              hipStream_t stream) {
  const float* x   = (const float*)d_in[0];
  const float* gw  = (const float*)d_in[1];
  const float* gb  = (const float*)d_in[2];
  const float* Wgu = (const float*)d_in[3];
  const float* Wd  = (const float*)d_in[4];
  float* out = (float*)d_out;

  if (ws_size < WS_NEED) return;   // workspace requirement (~228 MB)

  char* ws = (char*)d_ws;
  __bf16* xbf  = (__bf16*)(ws + WS_XBF);
  __bf16* wgut = (__bf16*)(ws + WS_WGUT);
  __bf16* wdt  = (__bf16*)(ws + WS_WDT);
  float*  ybuf = (float*)(ws + WS_YB);
  int*    cnt  = (int*)(ws + WS_CNT);
  int*    tok  = (int*)(ws + WS_TOK);
  float*  tw   = (float*)(ws + WS_TW);

  (void)in_sizes; (void)n_in; (void)out_size;

  int vec_blocks = (T_TOK * H_DIM / 4) / 256;                      // 4096
  k_zero_cnt<<<1, 32, 0, stream>>>(cnt);
  k_cvt   <<<vec_blocks, 256, 0, stream>>>(x, xbf);
  k_tconv <<<dim3(TWO_I / 32, H_DIM / 32, N_EXP), 256, 0, stream>>>(Wgu, wgut, H_DIM, TWO_I);
  k_tconv <<<dim3(H_DIM / 32, I_DIM / 32, N_EXP), 256, 0, stream>>>(Wd,  wdt,  I_DIM, H_DIM);
  k_router<<<T_TOK / 16, 256, 0, stream>>>(x, gw, gb, cnt, tok, tw);
  k_moe_mlp<<<N_EXP * (T_TOK / TILE_M), 256, 0, stream>>>(xbf, wgut, wdt, cnt, tok, tw, ybuf);
  k_comb  <<<vec_blocks, 256, 0, stream>>>(ybuf, out);
}
